// GCRNN_38929583571267
// MI455X (gfx1250) — compile-verified, matrix operand order checked by reading, not run on hardware
//
#include <hip/hip_runtime.h>

// ---------------- problem constants (from reference) ----------------
#define NNODES   200000
#define EMB      128
#define TSTEPS   32
#define NEDGES   100000
#define NACT     4096
#define NPAIR    2048
#define NCOMP    49998
#define UBASE    150000     // UID_MAX + 1
#define GDIM     512        // 4*EMB

// ---------------- WMMA types ----------------
typedef __attribute__((ext_vector_type(16))) __bf16 v16bf;
typedef __attribute__((ext_vector_type(8)))  float  v8f;

union ABFrag { uint4 q[2]; v16bf v; };

// Load one 16-lane bf16 fragment row: two 16B chunks at k0 and k0+16
// (A/B layout for V_WMMA_F32_16X16X32_BF16, ISA 7.12.2)
__device__ __forceinline__ v16bf load_frag(const unsigned short* __restrict__ base,
                                           int row, int k0) {
  const uint4* p0 = reinterpret_cast<const uint4*>(base + (size_t)row * EMB + k0);
  const uint4* p1 = reinterpret_cast<const uint4*>(base + (size_t)row * EMB + k0 + 16);
  ABFrag f; f.q[0] = *p0; f.q[1] = *p1; return f.v;
}

__device__ __forceinline__ unsigned short f2bf(float f) {  // RNE f32 -> bf16 bits
  unsigned u = __float_as_uint(f);
  unsigned r = u + 0x7FFFu + ((u >> 16) & 1u);
  return (unsigned short)(r >> 16);
}

// ---------------- utility kernels ----------------
__global__ void zero_f32(float* __restrict__ p, long n) {
  long i = (long)blockIdx.x * blockDim.x + threadIdx.x;
  long stride = (long)gridDim.x * blockDim.x;
  for (; i < n; i += stride) p[i] = 0.0f;
}

__global__ void cvt_bf16(const float* __restrict__ src, unsigned short* __restrict__ dst, long n) {
  long i = (long)blockIdx.x * blockDim.x + threadIdx.x;
  long stride = (long)gridDim.x * blockDim.x;
  for (; i < n; i += stride) dst[i] = f2bf(src[i]);
}

// ---------------- per-timestep kernels ----------------
// gather pre-aggregation h (as bf16) and c (f32) for active nodes
__global__ void __launch_bounds__(256)
gather_hc(const float* __restrict__ emb, const float* __restrict__ cx,
          const int* __restrict__ act, unsigned short* __restrict__ h_bf,
          float* __restrict__ prev_c) {
  int a = blockIdx.x * 8 + (threadIdx.x >> 5);
  int lane = threadIdx.x & 31;
  int node = act[a];
  const float4 h = *(const float4*)(emb + (size_t)node * EMB + lane * 4);
  const float4 c = *(const float4*)(cx  + (size_t)node * EMB + lane * 4);
  *(ushort4*)(h_bf + (size_t)a * EMB + lane * 4) =
      make_ushort4(f2bf(h.x), f2bf(h.y), f2bf(h.z), f2bf(h.w));
  *(float4*)(prev_c + (size_t)a * EMB + lane * 4) = c;
}

__global__ void __launch_bounds__(256)
gather_x(const float* __restrict__ emb, const int* __restrict__ act,
         unsigned short* __restrict__ x_bf) {
  int a = blockIdx.x * 8 + (threadIdx.x >> 5);
  int lane = threadIdx.x & 31;
  int node = act[a];
  const float4 x = *(const float4*)(emb + (size_t)node * EMB + lane * 4);
  *(ushort4*)(x_bf + (size_t)a * EMB + lane * 4) =
      make_ushort4(f2bf(x.x), f2bf(x.y), f2bf(x.z), f2bf(x.w));
}

// one wave per edge; 4 f32 atomics per lane -> global_atomic_add_f32
__global__ void __launch_bounds__(256)
scatter_edges(const float* __restrict__ emb, const int* __restrict__ src,
              const int* __restrict__ dst, float* __restrict__ msg,
              float* __restrict__ cnt) {
  int e = blockIdx.x * 8 + (threadIdx.x >> 5);
  if (e >= NEDGES) return;
  int lane = threadIdx.x & 31;
  int sN = src[e], dN = dst[e];
  const float4 v = *(const float4*)(emb + (size_t)sN * EMB + lane * 4);
  float* mp = msg + (size_t)dN * EMB + lane * 4;
  __hip_atomic_fetch_add(mp + 0, v.x, __ATOMIC_RELAXED, __HIP_MEMORY_SCOPE_AGENT);
  __hip_atomic_fetch_add(mp + 1, v.y, __ATOMIC_RELAXED, __HIP_MEMORY_SCOPE_AGENT);
  __hip_atomic_fetch_add(mp + 2, v.z, __ATOMIC_RELAXED, __HIP_MEMORY_SCOPE_AGENT);
  __hip_atomic_fetch_add(mp + 3, v.w, __ATOMIC_RELAXED, __HIP_MEMORY_SCOPE_AGENT);
  if (lane == 0)
    __hip_atomic_fetch_add(cnt + dN, 1.0f, __ATOMIC_RELAXED, __HIP_MEMORY_SCOPE_AGENT);
}

// emb[n] += msg[n]/cnt[n] for touched rows; re-zero msg/cnt (keeps invariant
// so we never re-memset 102MB between steps)
__global__ void __launch_bounds__(256)
update_nodes(float* __restrict__ emb, float* __restrict__ msg, float* __restrict__ cnt) {
  int n = blockIdx.x * 8 + (threadIdx.x >> 5);
  if (n >= NNODES) return;
  int lane = threadIdx.x & 31;
  float c = cnt[n];
  if (c <= 0.0f) return;
  float inv = 1.0f / c;
  size_t off = (size_t)n * EMB + lane * 4;
  float4 mv = *(float4*)(msg + off);
  float4 ev = *(float4*)(emb + off);
  ev.x += mv.x * inv; ev.y += mv.y * inv; ev.z += mv.z * inv; ev.w += mv.w * inv;
  *(float4*)(emb + off) = ev;
  float4 z = make_float4(0.f, 0.f, 0.f, 0.f);
  *(float4*)(msg + off) = z;
  if (lane == 0) cnt[n] = 0.0f;
}

// gates[4096,512] = x@W_ih^T + h@W_hh^T via bf16 WMMA, f32 accumulate.
// One wave per 16x16 output tile; 8 v_wmma per tile.
__global__ void __launch_bounds__(256)
gates_gemm(const unsigned short* __restrict__ x_bf,
           const unsigned short* __restrict__ h_bf,
           const unsigned short* __restrict__ Wihb,
           const unsigned short* __restrict__ Whhb,
           float* __restrict__ gates) {
  const int lane = threadIdx.x & 31;
  const int wave = threadIdx.x >> 5;
  const int tile = blockIdx.x * 8 + wave;     // 8192 tiles
  const int mT = tile >> 5;                    // 256 row-tiles
  const int nT = tile & 31;                    // 32 col-tiles
  const int half = lane >> 4;
  const int l16 = lane & 15;
  const int arow = mT * 16 + l16;
  const int brow = nT * 16 + l16;              // W row == gate column
  v8f acc = {};
#pragma unroll
  for (int kt = 0; kt < 4; ++kt) {
    int k0 = kt * 32 + half * 8;
    v16bf ax = load_frag(x_bf, arow, k0);
    v16bf bw = load_frag(Wihb, brow, k0);
    acc = __builtin_amdgcn_wmma_f32_16x16x32_bf16(false, ax, false, bw,
                                                  (short)0, acc, false, false);
    v16bf ah = load_frag(h_bf, arow, k0);
    v16bf bh = load_frag(Whhb, brow, k0);
    acc = __builtin_amdgcn_wmma_f32_16x16x32_bf16(false, ah, false, bh,
                                                  (short)0, acc, false, false);
  }
#pragma unroll
  for (int r = 0; r < 8; ++r) {
    int row = mT * 16 + r + 8 * half;
    gates[(size_t)row * GDIM + nT * 16 + l16] = acc[r];
  }
}

__global__ void __launch_bounds__(256)
lstm_pointwise(const float* __restrict__ gates, const float* __restrict__ b_ih,
               const float* __restrict__ b_hh, const float* __restrict__ prev_c,
               const int* __restrict__ act, float* __restrict__ emb,
               float* __restrict__ cx) {
  int idx = blockIdx.x * 256 + threadIdx.x;    // NACT*EMB threads
  int a = idx >> 7, d = idx & 127;
  const float* g = gates + (size_t)a * GDIM;
  float gi = g[d]         + b_ih[d]         + b_hh[d];
  float gf = g[128 + d]   + b_ih[128 + d]   + b_hh[128 + d];
  float gg = g[256 + d]   + b_ih[256 + d]   + b_hh[256 + d];
  float go = g[384 + d]   + b_ih[384 + d]   + b_hh[384 + d];
  float si = 1.0f / (1.0f + __expf(-gi));
  float sf = 1.0f / (1.0f + __expf(-gf));
  float so = 1.0f / (1.0f + __expf(-go));
  float cN = sf * prev_c[idx] + si * tanhf(gg);
  float hN = so * tanhf(cN);
  int node = act[a];
  emb[(size_t)node * EMB + d] = hN;
  cx[(size_t)node * EMB + d]  = cN;
}

// harvest u_emb[b] at its own timestep (replaces the [T,B,D] hist tensor)
__global__ void __launch_bounds__(256)
snapshot(const float* __restrict__ emb, const int* __restrict__ seed_users,
         const int* __restrict__ seed_time, float* __restrict__ u_emb, int t) {
  int b = blockIdx.x * 8 + (threadIdx.x >> 5);
  if (seed_time[b] != t) return;
  int lane = threadIdx.x & 31;
  *(float4*)(u_emb + (size_t)b * EMB + lane * 4) =
      *(const float4*)(emb + (size_t)seed_users[b] * EMB + lane * 4);
}

// fused logits GEMM + online logsumexp. One WG owns 16 rows, streams all
// column tiles (all_c stays L2-resident: 12.8MB bf16 << 192MB).
__global__ void __launch_bounds__(256)
lse_gemm(const unsigned short* __restrict__ u_bf,
         const unsigned short* __restrict__ allc_bf,   // 50000 rows x 128
         float* __restrict__ lse) {
  __shared__ float sm[8][16];
  __shared__ float ss[8][16];
  const int lane = threadIdx.x & 31;
  const int wave = threadIdx.x >> 5;
  const int half = lane >> 4;
  const int l16 = lane & 15;
  const int mT = blockIdx.x;                   // 128 row-tiles
  const int arow = mT * 16 + l16;
  float m[8], s[8];
#pragma unroll
  for (int r = 0; r < 8; ++r) { m[r] = -3.0e38f; s[r] = 0.0f; }
  const int NT = (NCOMP + 15) / 16;            // 3125
  for (int nT = wave; nT < NT; nT += 8) {
    if (nT + 8 < NT)
      __builtin_prefetch(allc_bf + (size_t)(nT + 8) * 16 * EMB, 0, 1);
    const int brow = nT * 16 + l16;
    v8f acc = {};
#pragma unroll
    for (int kt = 0; kt < 4; ++kt) {
      int k0 = kt * 32 + half * 8;
      v16bf a = load_frag(u_bf, arow, k0);
      v16bf b = load_frag(allc_bf, brow, k0);
      acc = __builtin_amdgcn_wmma_f32_16x16x32_bf16(false, a, false, b,
                                                    (short)0, acc, false, false);
    }
    int col = nT * 16 + l16;
    if (col < NCOMP) {
#pragma unroll
      for (int r = 0; r < 8; ++r) {
        float v = acc[r];
        float mn = fmaxf(m[r], v);
        s[r] = s[r] * __expf(m[r] - mn) + __expf(v - mn);
        m[r] = mn;
      }
    }
  }
  // combine across the 16 lanes of each half-wave (rows live in one half)
#pragma unroll
  for (int off = 1; off < 16; off <<= 1) {
#pragma unroll
    for (int r = 0; r < 8; ++r) {
      float mo = __shfl_xor(m[r], off, 32);
      float so = __shfl_xor(s[r], off, 32);
      float mn = fmaxf(m[r], mo);
      s[r] = s[r] * __expf(m[r] - mn) + so * __expf(mo - mn);
      m[r] = mn;
    }
  }
  if (l16 == 0) {
#pragma unroll
    for (int r = 0; r < 8; ++r) { sm[wave][r + 8 * half] = m[r]; ss[wave][r + 8 * half] = s[r]; }
  }
  __syncthreads();
  if (threadIdx.x < 16) {
    float M = -3.0e38f, S = 0.0f;
    for (int w = 0; w < 8; ++w) {
      float mo = sm[w][threadIdx.x], so = ss[w][threadIdx.x];
      float mn = fmaxf(M, mo);
      S = S * __expf(M - mn) + so * __expf(mo - mn);
      M = mn;
    }
    lse[mT * 16 + threadIdx.x] = M + __logf(S);
  }
}

__global__ void __launch_bounds__(256)
pos_kernel(const float* __restrict__ u_emb, const float* __restrict__ ent_emb,
           const int* __restrict__ rel, float* __restrict__ pos) {
  int b = blockIdx.x * 8 + (threadIdx.x >> 5);
  int lane = threadIdx.x & 31;
  const float4 u = *(const float4*)(u_emb + (size_t)b * EMB + lane * 4);
  const float4 c = *(const float4*)(ent_emb + (size_t)(UBASE + rel[b]) * EMB + lane * 4);
  float d = u.x * c.x + u.y * c.y + u.z * c.z + u.w * c.w;
  for (int off = 16; off; off >>= 1) d += __shfl_down(d, off, 32);
  if (lane == 0) pos[b] = d;
}

__global__ void __launch_bounds__(256)
loss_kernel(const float* __restrict__ lse, const float* __restrict__ pos,
            float* __restrict__ out) {
  __shared__ float sh[256];
  float acc = 0.0f;
  for (int b = threadIdx.x; b < NPAIR; b += 256) acc += lse[b] - pos[b];
  sh[threadIdx.x] = acc;
  __syncthreads();
  for (int off = 128; off; off >>= 1) {
    if ((int)threadIdx.x < off) sh[threadIdx.x] += sh[threadIdx.x + off];
    __syncthreads();
  }
  if (threadIdx.x == 0) out[0] = sh[0];
}

// ---------------- host orchestration ----------------
static inline size_t alignUp(size_t x, size_t a) { return (x + a - 1) & ~(a - 1); }

extern "C" void kernel_launch(void* const* d_in, const int* in_sizes, int n_in,
                              void* d_out, int out_size, void* d_ws, size_t ws_size,
                              hipStream_t stream) {
  const float* ent_emb = (const float*)d_in[0];
  const float* c0_emb  = (const float*)d_in[1];
  const float* W_ih    = (const float*)d_in[2];
  const float* W_hh    = (const float*)d_in[3];
  const float* b_ih    = (const float*)d_in[4];
  const float* b_hh    = (const float*)d_in[5];
  const int* edges_src = (const int*)d_in[6];
  const int* edges_dst = (const int*)d_in[7];
  const int* act_idx   = (const int*)d_in[8];
  const int* seed_users= (const int*)d_in[9];
  const int* seed_time = (const int*)d_in[10];
  const int* comp_rel  = (const int*)d_in[11];
  float* out = (float*)d_out;

  char* base = (char*)d_ws;
  size_t off = 0;
  auto alloc = [&](size_t bytes) { size_t o = off; off = alignUp(off + bytes, 256); return o; };

  float* emb    = (float*)(base + alloc((size_t)NNODES * EMB * 4));
  float* cx     = (float*)(base + alloc((size_t)NNODES * EMB * 4));
  float* msg    = (float*)(base + alloc((size_t)NNODES * EMB * 4));
  float* cnt    = (float*)(base + alloc((size_t)NNODES * 4));
  unsigned short* Wihb   = (unsigned short*)(base + alloc((size_t)GDIM * EMB * 2));
  unsigned short* Whhb   = (unsigned short*)(base + alloc((size_t)GDIM * EMB * 2));
  unsigned short* x_bf   = (unsigned short*)(base + alloc((size_t)NACT * EMB * 2));
  unsigned short* h_bf   = (unsigned short*)(base + alloc((size_t)NACT * EMB * 2));
  float* prev_c = (float*)(base + alloc((size_t)NACT * EMB * 4));
  float* gates  = (float*)(base + alloc((size_t)NACT * GDIM * 4));
  float* u_emb  = (float*)(base + alloc((size_t)NPAIR * EMB * 4));
  unsigned short* u_bf   = (unsigned short*)(base + alloc((size_t)NPAIR * EMB * 2));
  unsigned short* allc_bf= (unsigned short*)(base + alloc((size_t)50000 * EMB * 2)); // 2 pad rows (real ids 199998/199999)
  float* lse    = (float*)(base + alloc((size_t)NPAIR * 4));
  float* pos    = (float*)(base + alloc((size_t)NPAIR * 4));

  // init state
  hipMemcpyAsync(emb, ent_emb, (size_t)NNODES * EMB * 4, hipMemcpyDeviceToDevice, stream);
  hipMemcpyAsync(cx,  c0_emb,  (size_t)NNODES * EMB * 4, hipMemcpyDeviceToDevice, stream);
  zero_f32<<<4096, 256, 0, stream>>>(msg, (long)NNODES * EMB);
  zero_f32<<<1024, 256, 0, stream>>>(cnt, (long)NNODES);
  cvt_bf16<<<256, 256, 0, stream>>>(W_ih, Wihb, (long)GDIM * EMB);
  cvt_bf16<<<256, 256, 0, stream>>>(W_hh, Whhb, (long)GDIM * EMB);
  cvt_bf16<<<2048, 256, 0, stream>>>(ent_emb + (size_t)UBASE * EMB, allc_bf, (long)50000 * EMB);

  for (int t = 0; t < TSTEPS; ++t) {
    const int* src = edges_src + (size_t)t * NEDGES;
    const int* dst = edges_dst + (size_t)t * NEDGES;
    const int* act = act_idx   + (size_t)t * NACT;
    gather_hc<<<NACT / 8, 256, 0, stream>>>(emb, cx, act, h_bf, prev_c);
    scatter_edges<<<NEDGES / 8, 256, 0, stream>>>(emb, src, dst, msg, cnt);
    update_nodes<<<NNODES / 8, 256, 0, stream>>>(emb, msg, cnt);
    gather_x<<<NACT / 8, 256, 0, stream>>>(emb, act, x_bf);
    gates_gemm<<<(NACT / 16) * (GDIM / 16) / 8, 256, 0, stream>>>(x_bf, h_bf, Wihb, Whhb, gates);
    lstm_pointwise<<<(NACT * EMB) / 256, 256, 0, stream>>>(gates, b_ih, b_hh, prev_c, act, emb, cx);
    snapshot<<<NPAIR / 8, 256, 0, stream>>>(emb, seed_users, seed_time, u_emb, t);
  }

  cvt_bf16<<<256, 256, 0, stream>>>(u_emb, u_bf, (long)NPAIR * EMB);
  lse_gemm<<<NPAIR / 16, 256, 0, stream>>>(u_bf, allc_bf, lse);
  pos_kernel<<<NPAIR / 8, 256, 0, stream>>>(u_emb, ent_emb, comp_rel, pos);
  loss_kernel<<<1, 256, 0, stream>>>(lse, pos, out);
}
